// Net_51994874085376
// MI455X (gfx1250) — matrix operand order repaired and checked by reading, not compiled, not run on hardware
//
#include <hip/hip_runtime.h>
#include <hip/hip_bf16.h>

typedef __attribute__((ext_vector_type(16))) _Float16 v16h;
typedef __attribute__((ext_vector_type(8)))  _Float16 v8h;
typedef __attribute__((ext_vector_type(8)))  float    v8f;

#define NWG   64
#define TPB   256
#define NTHREADS (NWG * TPB)

#define Bsz   128
#define Tsz   64
#define NIN   2048
#define HID1  2048
#define HID2  1024
#define NOUT  128
#define BETA  0.95f
#define THRESH 1.0f

#define SHUF16(lo, hi) __builtin_shufflevector(lo, hi, 0,1,2,3,4,5,6,7,8,9,10,11,12,13,14,15)

// -------- grid-wide barrier: monotonic counter, zeroed per launch ----------
__device__ inline void grid_sync(volatile unsigned* cnt, unsigned target) {
  __syncthreads();
  if (threadIdx.x == 0) {
    __threadfence();                 // release: make spike stores visible
    atomicAdd((unsigned*)cnt, 1u);
    while (*cnt < target) {
      __builtin_amdgcn_s_sleep(2);
    }
    __threadfence();                 // acquire: see other WGs' stores
  }
  __syncthreads();
}

// -------- cooperative flat copy of a contiguous f16 slab into LDS ----------
__device__ inline void stage(const _Float16* __restrict__ src, _Float16* dst,
                             int nhalves) {
  for (int i = threadIdx.x * 8; i < nhalves; i += TPB * 8) {
    *(v8h*)(dst + i) = *(const v8h*)(src + i);
  }
}

// -------- stage two 16-row m-slabs row-interleaved: A-row r -> 2r*K,
//          B-row r -> (2r+1)*K  (keeps all DS offsets within immediates) ----
__device__ inline void stage_pair(const _Float16* __restrict__ srcA,
                                  const _Float16* __restrict__ srcB,
                                  _Float16* dst, int K, int logK) {
  const int total = 16 * K / 8;                 // 16B chunks per slab
  for (int i = threadIdx.x; i < total; i += TPB) {
    int off = i * 8;
    int row = off >> logK;
    int col = off & (K - 1);
    *(v8h*)(dst + (size_t)(2 * row) * K + col)     = *(const v8h*)(srcA + off);
    *(v8h*)(dst + (size_t)(2 * row + 1) * K + col) = *(const v8h*)(srcB + off);
  }
}

// Fragment layouts (wave32, V_WMMA_F32_16X16X32_F16):
// A (16x32): lane holds row m=lane%16, K = {kb..kb+7, kb+16..kb+23}, kb=8*(lane>=16)
// B (32x16): lane holds W-row n=lane%16, K = {kb..kb+7, kb+16..kb+23}, kb=8*(lane>=16)
// C/D:       lane holds n=lane%16, rows m = i + 8*(lane>=16), i=0..7

// -------- two 16x16 tiles (row-interleaved m-slabs in LDS, shared B) -------
// Software-pipelined: loads for k+32 issue before the WMMAs consuming k.
__device__ inline void gemm_pair_lds(const _Float16* ldsAB,
                                     const _Float16* __restrict__ Wg,
                                     const float* __restrict__ bias,
                                     int K, int nt, int lane,
                                     v8f* outA, v8f* outB) {
  const int l16 = lane & 15;
  const int hi  = lane >> 4;
  const _Float16* ap = ldsAB + (size_t)(2 * l16) * K + hi * 8;  // B row at +K
  const _Float16* bp = Wg + (size_t)(nt * 16 + l16) * K + hi * 8;
  const float bv = bias[nt * 16 + l16];
  v8f accA = {bv, bv, bv, bv, bv, bv, bv, bv};
  v8f accB = accA;

  v8h wb0 = *(const v8h*)(bp + 0);
  v8h wb1 = *(const v8h*)(bp + 16);
  v8h a0  = *(const v8h*)(ap + 0);
  v8h a1  = *(const v8h*)(ap + 16);
  v8h e0  = *(const v8h*)(ap + K + 0);
  v8h e1  = *(const v8h*)(ap + K + 16);

  for (int k = 0; k < K; k += 32) {
    v16h bf = SHUF16(wb0, wb1);
    v16h af = SHUF16(a0, a1);
    v16h ef = SHUF16(e0, e1);
    const int kn = k + 32;
    if (kn < K) {                    // prefetch next iteration's fragments
      wb0 = *(const v8h*)(bp + kn);
      wb1 = *(const v8h*)(bp + kn + 16);
      a0  = *(const v8h*)(ap + kn);
      a1  = *(const v8h*)(ap + kn + 16);
      e0  = *(const v8h*)(ap + K + kn);
      e1  = *(const v8h*)(ap + K + kn + 16);
    }
    accA = __builtin_amdgcn_wmma_f32_16x16x32_f16(false, af, false, bf, (short)0,
                                                  accA, false, false);
    accB = __builtin_amdgcn_wmma_f32_16x16x32_f16(false, ef, false, bf, (short)0,
                                                  accB, false, false);
  }
  *outA = accA;
  *outB = accB;
}

// -------- one 16x16 tile, contiguous A slab in LDS (pipelined) -------------
__device__ inline v8f gemm_one_lds(const _Float16* ldsA,
                                   const _Float16* __restrict__ Wg,
                                   const float* __restrict__ bias,
                                   int K, int nt, int lane) {
  const int l16 = lane & 15;
  const int hi  = lane >> 4;
  const _Float16* ap = ldsA + (size_t)l16 * K + hi * 8;
  const _Float16* bp = Wg + (size_t)(nt * 16 + l16) * K + hi * 8;
  const float bv = bias[nt * 16 + l16];
  v8f acc = {bv, bv, bv, bv, bv, bv, bv, bv};

  v8h wb0 = *(const v8h*)(bp + 0);
  v8h wb1 = *(const v8h*)(bp + 16);
  v8h a0  = *(const v8h*)(ap + 0);
  v8h a1  = *(const v8h*)(ap + 16);

  for (int k = 0; k < K; k += 32) {
    v16h bf = SHUF16(wb0, wb1);
    v16h af = SHUF16(a0, a1);
    const int kn = k + 32;
    if (kn < K) {
      wb0 = *(const v8h*)(bp + kn);
      wb1 = *(const v8h*)(bp + kn + 16);
      a0  = *(const v8h*)(ap + kn);
      a1  = *(const v8h*)(ap + kn + 16);
    }
    acc = __builtin_amdgcn_wmma_f32_16x16x32_f16(false, af, false, bf, (short)0,
                                                 acc, false, false);
  }
  return acc;
}

// -------- LIF (subtract reset) + f16 spike store for next layer ------------
__device__ inline void lif_spike(v8f& mem, const v8f cur, _Float16* __restrict__ Sout,
                                 int N, int mt, int nt, int lane) {
  const int l16 = lane & 15;
  const int mb  = mt * 16 + ((lane >> 4) << 3);
  const int n   = nt * 16 + l16;
#pragma unroll
  for (int i = 0; i < 8; ++i) {
    float mo = mem[i];
    float mn = BETA * mo + cur[i] - ((mo > THRESH) ? THRESH : 0.0f);
    mem[i] = mn;
    Sout[(size_t)(mb + i) * N + n] = (_Float16)((mn > THRESH) ? 1.0f : 0.0f);
  }
}

// -------- LIF for output layer: write spk3/mem3 f32 to d_out ---------------
__device__ inline void lif_out(v8f& mem, const v8f cur, float* __restrict__ spk_out,
                               float* __restrict__ mem_out, int t, int mt, int nt,
                               int lane) {
  const int l16 = lane & 15;
  const int mb  = mt * 16 + ((lane >> 4) << 3);
  const int n   = nt * 16 + l16;
#pragma unroll
  for (int i = 0; i < 8; ++i) {
    float mo = mem[i];
    float mn = BETA * mo + cur[i] - ((mo > THRESH) ? THRESH : 0.0f);
    mem[i] = mn;
    size_t off = ((size_t)(mb + i) * Tsz + t) * NOUT + n;
    spk_out[off] = (mn > THRESH) ? 1.0f : 0.0f;
    mem_out[off] = mn;
  }
}

__global__ __launch_bounds__(TPB) void snn_persistent(
    const float* __restrict__ x,
    const float* __restrict__ W0, const float* __restrict__ b0,
    const float* __restrict__ W1, const float* __restrict__ b1,
    const float* __restrict__ W2, const float* __restrict__ b2,
    const float* __restrict__ W3, const float* __restrict__ b3,
    float* __restrict__ out, void* __restrict__ ws) {

  // two 16x2048 f16 m-slabs, row-interleaved; 128 KB of the 320 KB WGP LDS
  __shared__ __align__(16) _Float16 lds[2 * 16 * 2048];

  volatile unsigned* cnt = (volatile unsigned*)ws;
  _Float16* base = (_Float16*)((char*)ws + 256);
  _Float16* W0h = base;                                 // 2048*2048
  _Float16* W1h = W0h + (size_t)NIN * NIN;              // 2048*2048
  _Float16* W2h = W1h + (size_t)HID1 * NIN;             // 1024*2048
  _Float16* W3h = W2h + (size_t)HID2 * HID1;            // 128*1024
  _Float16* s0  = W3h + (size_t)NOUT * HID2;            // 128*2048
  _Float16* s1  = s0  + (size_t)Bsz * NIN;              // 128*2048
  _Float16* s2  = s1  + (size_t)Bsz * HID1;             // 128*1024
  _Float16* xh  = s2  + (size_t)Bsz * HID2;             // 128*2048 staging (f16)

  const int tid  = blockIdx.x * TPB + threadIdx.x;
  const int lane = threadIdx.x & 31;
  const int gw   = tid >> 5;                            // global wave id, 0..511

  // ---------------- prologue: f32 -> f16 weight conversion + x slab t=0 ----
  for (size_t i = tid; i < (size_t)NIN * NIN;  i += NTHREADS) W0h[i] = (_Float16)W0[i];
  for (size_t i = tid; i < (size_t)HID1 * NIN; i += NTHREADS) W1h[i] = (_Float16)W1[i];
  for (size_t i = tid; i < (size_t)HID2 * HID1;i += NTHREADS) W2h[i] = (_Float16)W2[i];
  for (size_t i = tid; i < (size_t)NOUT * HID2;i += NTHREADS) W3h[i] = (_Float16)W3[i];
  for (int i = tid; i < Bsz * NIN; i += NTHREADS) {
    int m = i >> 11, k = i & (NIN - 1);
    xh[i] = (_Float16)x[(size_t)m * Tsz * NIN + k];     // x[m, 0, k]
  }
  unsigned bar = 0;
  grid_sync(cnt, ++bar * NWG);

  // ---------------- static tile ownership (m-tiles are WG-uniform) ---------
  const int tA_m = blockIdx.x >> 4;        // L0/L1 first m-tile   (0..3)
  const int tB_m = tA_m + 4;               // L0/L1 second m-tile  (4..7)
  const int tAB_n = gw & 127;              // shared n-tile        (0..127)
  const int t2_m = blockIdx.x >> 3;        // L2 m-tile            (0..7)
  const int t2_n = gw & 63;                // L2 n-tile            (0..63)
  const int t3_m = blockIdx.x;             // L3 m-tile (WGs 0..7)
  const int t3_n = threadIdx.x >> 5;       // L3 n-tile (wave in WG, 0..7)

  // membrane state lives in VGPRs for the whole run
  v8f m0a = {}, m0b = {}, m1a = {}, m1b = {}, m2 = {}, m3 = {};

  float* spk_out = out;
  float* mem_out = out + (size_t)Bsz * Tsz * NOUT;

  for (int t = 0; t < Tsz; ++t) {
    // ---- layer 0: cur0 = xh @ W0^T + b0 ; spikes -> s0 --------------------
    {
      stage_pair(xh + (size_t)tA_m * 16 * NIN, xh + (size_t)tB_m * 16 * NIN,
                 lds, NIN, 11);
      __syncthreads();
      v8f cA, cB;
      gemm_pair_lds(lds, W0h, b0, NIN, tAB_n, lane, &cA, &cB);
      lif_spike(m0a, cA, s0, NIN, tA_m, tAB_n, lane);
      lif_spike(m0b, cB, s0, NIN, tB_m, tAB_n, lane);
    }
    grid_sync(cnt, ++bar * NWG);

    // ---- layer 1: cur1 = s0 @ W1^T + b1 ; spikes -> s1 --------------------
    {
      stage_pair(s0 + (size_t)tA_m * 16 * NIN, s0 + (size_t)tB_m * 16 * NIN,
                 lds, NIN, 11);
      __syncthreads();
      v8f cA, cB;
      gemm_pair_lds(lds, W1h, b1, NIN, tAB_n, lane, &cA, &cB);
      lif_spike(m1a, cA, s1, HID1, tA_m, tAB_n, lane);
      lif_spike(m1b, cB, s1, HID1, tB_m, tAB_n, lane);
    }
    grid_sync(cnt, ++bar * NWG);

    // ---- layer 2: cur2 = s1 @ W2^T + b2 ; spikes -> s2 --------------------
    {
      stage(s1 + (size_t)t2_m * 16 * HID1, lds, 16 * HID1);
      __syncthreads();
      v8f c = gemm_one_lds(lds, W2h, b2, HID1, t2_n, lane);
      lif_spike(m2, c, s2, HID2, t2_m, t2_n, lane);
    }
    grid_sync(cnt, ++bar * NWG);

    // ---- layer 3 on WGs 0..7; other WGs stage the next x slab -------------
    if (blockIdx.x < 8) {
      stage(s2 + (size_t)t3_m * 16 * HID2, lds, 16 * HID2);
      __syncthreads();
      v8f c = gemm_one_lds(lds, W3h, b3, HID2, t3_n, lane);
      lif_out(m3, c, spk_out, mem_out, t, t3_m, t3_n, lane);
    } else if (t + 1 < Tsz) {
      const int stride = NTHREADS - 8 * TPB;
      int tid2 = tid - 8 * TPB;
      for (int i = tid2; i < Bsz * NIN; i += stride) {
        int m = i >> 11, k = i & (NIN - 1);
        xh[i] = (_Float16)x[((size_t)m * Tsz + (t + 1)) * NIN + k];
      }
    }
    grid_sync(cnt, ++bar * NWG);
  }
}

extern "C" void kernel_launch(void* const* d_in, const int* in_sizes, int n_in,
                              void* d_out, int out_size, void* d_ws, size_t ws_size,
                              hipStream_t stream) {
  const float* x  = (const float*)d_in[0];
  const float* W0 = (const float*)d_in[1];
  const float* b0 = (const float*)d_in[2];
  const float* W1 = (const float*)d_in[3];
  const float* b1 = (const float*)d_in[4];
  const float* W2 = (const float*)d_in[5];
  const float* b2 = (const float*)d_in[6];
  const float* W3 = (const float*)d_in[7];
  const float* b3 = (const float*)d_in[8];

  // reset the grid-barrier counter (capture-legal)
  hipMemsetAsync(d_ws, 0, 256, stream);

  snn_persistent<<<dim3(NWG), dim3(TPB), 0, stream>>>(
      x, W0, b0, W1, b1, W2, b2, W3, b3, (float*)d_out, d_ws);
}
